// LSTM_47854525612158
// MI455X (gfx1250) — matrix operand order repaired and checked by reading, compiled, not run on hardware
//
#include <hip/hip_runtime.h>

// ---------------------------------------------------------------------------
// LSTM for MI455X (gfx1250): bf16x3 split-product WMMA GEMMs, fused recurrence.
// N=64, T=512, D=1024, H=1024.
// Round 2: split K-loop (immediate-offset loads, no cndmask) + 3 independent
// accumulators (no WMMA->WMMA hazard NOPs).
// ---------------------------------------------------------------------------

typedef __attribute__((ext_vector_type(16))) __bf16 v16bf;
typedef __attribute__((ext_vector_type(8)))  __bf16 v8bf;
typedef __attribute__((ext_vector_type(8)))  float  v8f;

#define N_  64
#define T_  512
#define D_  1024
#define H_  1024
#define K_  (D_ + H_)          // 2048 combined contraction dim
#define KT_ 64                 // K/32 k-tiles (32 from x, 32 from h)

// bf16 round-to-nearest-even helpers (bit level)
__device__ __forceinline__ unsigned short f2bf_bits(float f) {
  unsigned u = __float_as_uint(f);
  u += 0x7fffu + ((u >> 16) & 1u);
  return (unsigned short)(u >> 16);
}
__device__ __forceinline__ float bf2f(unsigned short b) {
  return __uint_as_float(((unsigned)b) << 16);
}

// ---------------------------------------------------------------------------
// Pack [Wx ; Wh] (2048 x 4096, row-major f32) into WMMA B-operand order,
// split into bf16 hi/lo. Element p = (((nt*KT + kt)*32 + lane)*16 + e):
//   col = nt*16 + (lane&15);  k = kt*32 + (lane>>4)*8 + (e&7) + (e>>3)*16   (ISA B layout)
// ---------------------------------------------------------------------------
__global__ void pack_weights(const float* __restrict__ Wx,
                             const float* __restrict__ Wh,
                             unsigned short* __restrict__ Whi,
                             unsigned short* __restrict__ Wlo) {
  int p = blockIdx.x * blockDim.x + threadIdx.x;   // exactly 2048*4096 threads
  int e    = p & 15;
  int lane = (p >> 4) & 31;
  int kt   = (p >> 9) & (KT_ - 1);
  int nt   = p >> 15;
  int k    = kt * 32 + ((lane >> 4) << 3) + (e & 7) + ((e >> 3) << 4);
  int col  = nt * 16 + (lane & 15);
  float w = (k < D_) ? Wx[(size_t)k * (4 * H_) + col]
                     : Wh[(size_t)(k - D_) * (4 * H_) + col];
  unsigned short hb = f2bf_bits(w);
  Whi[p] = hb;
  Wlo[p] = f2bf_bits(w - bf2f(hb));
}

// Split an f32 array into bf16 hi/lo arrays.
__global__ void split_f32(const float* __restrict__ src,
                          unsigned short* __restrict__ hi,
                          unsigned short* __restrict__ lo, int n) {
  int i = blockIdx.x * blockDim.x + threadIdx.x;
  if (i < n) {
    float v = src[i];
    unsigned short h = f2bf_bits(v);
    hi[i] = h;
    lo[i] = f2bf_bits(v - bf2f(h));
  }
}

__global__ void zero_f32(float* __restrict__ p, int n) {
  int i = blockIdx.x * blockDim.x + threadIdx.x;
  if (i < n) p[i] = 0.0f;
}

// ---------------------------------------------------------------------------
// One LSTM timestep:  a = [x_t | h_prev] @ [Wx;Wh] + b ; gates; c,h update.
// Grid: 64 blocks (16-column slice of H each). Block: 512 thr = 16 waves
//       = 4 M-tiles x 4 gates. Each wave: 16x16 f32 tile over K=2048.
// ---------------------------------------------------------------------------
__global__ __launch_bounds__(512) void lstm_step(
    const unsigned short* __restrict__ xhi, const unsigned short* __restrict__ xlo,
    const unsigned short* __restrict__ Whi, const unsigned short* __restrict__ Wlo,
    const unsigned short* __restrict__ hhi, const unsigned short* __restrict__ hlo,
    unsigned short* __restrict__ nhhi,      unsigned short* __restrict__ nhlo,
    float* __restrict__ c, const float* __restrict__ bias,
    float* __restrict__ out, int t)
{
  const int tid  = threadIdx.x;
  const int lane = tid & 31;
  const int wv   = tid >> 5;
  const int mt   = wv & 3;        // M tile (16 rows each, N=64 total)
  const int gate = wv >> 2;       // 0=i 1=f 2=o 3=g
  const int jb   = blockIdx.x;    // column slice of H
  const int nt   = gate * 64 + jb;

  // A-operand lane mapping (16-bit A 16x32): row = lane&15,
  // element e reads K = klo + e (e<8) / klo + 8 + e (e>=8), klo = (lane>>4)*8.
  const int rowA = mt * 16 + (lane & 15);
  const int klo  = (lane >> 4) << 3;

  const unsigned short* xr_hi = xhi + ((size_t)rowA * T_ + t) * D_ + klo;
  const unsigned short* xr_lo = xlo + ((size_t)rowA * T_ + t) * D_ + klo;
  const unsigned short* hr_hi = hhi + (size_t)rowA * H_ + klo;
  const unsigned short* hr_lo = hlo + (size_t)rowA * H_ + klo;
  const unsigned short* bw_hi = Whi + ((size_t)nt * KT_ * 32 + lane) * 16;
  const unsigned short* bw_lo = Wlo + ((size_t)nt * KT_ * 32 + lane) * 16;

  // Three independent accumulators: removes WMMA->WMMA RAW hazards inside the
  // k-loop (dependence distance = 3 WMMAs), summed once at the end.
  v8f acc0 = {}, acc1 = {}, acc2 = {};

  // ---- x_t part: k-tiles 0..31, all addresses = base + immediate offset ----
  #pragma unroll 8
  for (int kt = 0; kt < 32; ++kt) {
    union { v16bf v; v8bf h[2]; } Ah, Al;
    Ah.h[0] = *(const v8bf*)(xr_hi + kt * 32);
    Ah.h[1] = *(const v8bf*)(xr_hi + kt * 32 + 16);
    Al.h[0] = *(const v8bf*)(xr_lo + kt * 32);
    Al.h[1] = *(const v8bf*)(xr_lo + kt * 32 + 16);
    v16bf Bh = *(const v16bf*)(bw_hi + (size_t)kt * 512);
    v16bf Bl = *(const v16bf*)(bw_lo + (size_t)kt * 512);
    acc0 = __builtin_amdgcn_wmma_f32_16x16x32_bf16(false, Ah.v, false, Bh, (short)0, acc0, false, false);
    acc1 = __builtin_amdgcn_wmma_f32_16x16x32_bf16(false, Ah.v, false, Bl, (short)0, acc1, false, false);
    acc2 = __builtin_amdgcn_wmma_f32_16x16x32_bf16(false, Al.v, false, Bh, (short)0, acc2, false, false);
  }

  // ---- h_prev part: k-tiles 32..63 ----
  #pragma unroll 8
  for (int kt = 0; kt < 32; ++kt) {
    union { v16bf v; v8bf h[2]; } Ah, Al;
    Ah.h[0] = *(const v8bf*)(hr_hi + kt * 32);
    Ah.h[1] = *(const v8bf*)(hr_hi + kt * 32 + 16);
    Al.h[0] = *(const v8bf*)(hr_lo + kt * 32);
    Al.h[1] = *(const v8bf*)(hr_lo + kt * 32 + 16);
    v16bf Bh = *(const v16bf*)(bw_hi + (size_t)(32 + kt) * 512);
    v16bf Bl = *(const v16bf*)(bw_lo + (size_t)(32 + kt) * 512);
    acc0 = __builtin_amdgcn_wmma_f32_16x16x32_bf16(false, Ah.v, false, Bh, (short)0, acc0, false, false);
    acc1 = __builtin_amdgcn_wmma_f32_16x16x32_bf16(false, Ah.v, false, Bl, (short)0, acc1, false, false);
    acc2 = __builtin_amdgcn_wmma_f32_16x16x32_bf16(false, Al.v, false, Bh, (short)0, acc2, false, false);
  }

  // C/D layout: VGPR r, lane l -> M = r + 8*(l>>4), N = l&15
  __shared__ float sacc[4][N_][16];
  const int r0 = mt * 16 + ((lane >> 4) << 3);
  #pragma unroll
  for (int r = 0; r < 8; ++r)
    sacc[gate][r0 + r][lane & 15] = acc0[r] + (acc1[r] + acc2[r]);
  __syncthreads();

  // Gate math: 64 rows x 16 cols = 1024 elements over 512 threads
  for (int e = tid; e < N_ * 16; e += 512) {
    int n  = e >> 4;
    int jc = e & 15;
    int j  = jb * 16 + jc;
    float ai = sacc[0][n][jc] + bias[j];
    float af = sacc[1][n][jc] + bias[H_ + j];
    float ao = sacc[2][n][jc] + bias[2 * H_ + j];
    float ag = sacc[3][n][jc] + bias[3 * H_ + j];
    float ig = 1.0f / (1.0f + __expf(-ai));
    float fg = 1.0f / (1.0f + __expf(-af));
    float og = 1.0f / (1.0f + __expf(-ao));
    float gg = tanhf(ag);
    size_t cidx = (size_t)n * H_ + j;
    float cn = fg * c[cidx] + ig * gg;
    c[cidx] = cn;
    float hn = og * tanhf(cn);
    out[((size_t)n * T_ + t) * H_ + j] = hn;
    unsigned short hb = f2bf_bits(hn);
    nhhi[cidx] = hb;
    nhlo[cidx] = f2bf_bits(hn - bf2f(hb));
  }
}

// ---------------------------------------------------------------------------
// Workspace layout (bytes), total ~161 MB:
//   Whi  @ 0          (16,777,216)   Wlo @ 16,777,216 (16,777,216)
//   xhi  @ 33,554,432 (67,108,864)   xlo @ 100,663,296 (67,108,864)
//   hbh  @ 167,772,160 (2 x 64K bf16 = 262,144)   hbl @ +262,144
//   c    @ +524,288   (262,144)
// ---------------------------------------------------------------------------
extern "C" void kernel_launch(void* const* d_in, const int* in_sizes, int n_in,
                              void* d_out, int out_size, void* d_ws, size_t ws_size,
                              hipStream_t stream) {
  const float* x  = (const float*)d_in[0];   // (N,T,D)
  const float* h0 = (const float*)d_in[1];   // (N,H)
  const float* Wx = (const float*)d_in[2];   // (D,4H)
  const float* Wh = (const float*)d_in[3];   // (H,4H)
  const float* b  = (const float*)d_in[4];   // (4H)
  float* out = (float*)d_out;                // (N,T,H)

  char* ws = (char*)d_ws;
  unsigned short* Whi = (unsigned short*)(ws);
  unsigned short* Wlo = (unsigned short*)(ws + 16777216);
  unsigned short* xhi = (unsigned short*)(ws + 33554432);
  unsigned short* xlo = (unsigned short*)(ws + 100663296);
  unsigned short* hbh = (unsigned short*)(ws + 167772160);
  unsigned short* hbl = (unsigned short*)(ws + 167772160 + 262144);
  float*          c   = (float*)(ws + 167772160 + 524288);

  const int NXH = N_ * H_;                 // 65536
  const int NX  = N_ * T_ * D_;            // 33,554,432

  pack_weights<<<(K_ * 4 * H_) / 256, 256, 0, stream>>>(Wx, Wh, Whi, Wlo);
  split_f32<<<(NX + 255) / 256, 256, 0, stream>>>(x, xhi, xlo, NX);
  split_f32<<<(NXH + 255) / 256, 256, 0, stream>>>(h0, hbh, hbl, NXH);
  zero_f32<<<(NXH + 255) / 256, 256, 0, stream>>>(c, NXH);

  for (int t = 0; t < T_; ++t) {
    const unsigned short* hh = hbh + (size_t)(t & 1) * NXH;
    const unsigned short* hl = hbl + (size_t)(t & 1) * NXH;
    unsigned short* nh = hbh + (size_t)((t + 1) & 1) * NXH;
    unsigned short* nl = hbl + (size_t)((t + 1) & 1) * NXH;
    lstm_step<<<H_ / 16, 512, 0, stream>>>(xhi, xlo, Whi, Wlo,
                                           hh, hl, nh, nl, c, b, out, t);
  }
}
